// LSTMModel_19473381720745
// MI455X (gfx1250) — matrix operand order repaired and checked by reading, not compiled, not run on hardware
//
#include <hip/hip_runtime.h>
#include <hip/hip_bf16.h>

// Problem constants (from reference): B=64, T=512, D=H=1024, 4H=4096
#define B_    64
#define T_    512
#define D_    1024
#define H_    1024
#define G4_   4096
#define NBLK  32
#define TPB   256

typedef __bf16 bf16_t;
typedef __attribute__((ext_vector_type(16))) __bf16 bf16x16;
typedef __attribute__((ext_vector_type(8)))  float  f32x8;
typedef unsigned int u32x4 __attribute__((ext_vector_type(4)));
typedef int          i32x4 __attribute__((ext_vector_type(4)));
typedef int          i32x8 __attribute__((ext_vector_type(8)));

// ---------------------------------------------------------------------------
// Tensor Data Mover: contiguous `bytes` (multiple of 8, here 32 KB) copy
// global -> LDS as a 1-row tile of 8-byte elements (D# per ISA ch.8).
// Issued once per wave that executes it (EXEC ignored); caller guards to one
// wave per block and waits on TENSORcnt before the workgroup barrier.
// ---------------------------------------------------------------------------
__device__ __forceinline__ void tdm_load_to_lds(void* lds_dst, const void* gsrc,
                                                unsigned bytes) {
  const unsigned elems = bytes >> 3;                       // data_size = 8B units
  const unsigned long long ga = (unsigned long long)(uintptr_t)gsrc;
  const unsigned la = (unsigned)(uintptr_t)lds_dst;        // low 32b == LDS offset
  u32x4 g0 = { 1u,                                         // count=1 (valid D#)
               la,                                         // lds_addr
               (unsigned)ga,                               // global_addr[31:0]
               (unsigned)(ga >> 32) | (2u << 30) };        // ga[56:32] | type=2
  i32x8 g1 = { (int)(3u << 16),                            // wg_mask=0, data_size=3(8B)
               (int)((elems & 0xFFFFu) << 16),             // tensor_dim0[15:0]
               (int)((elems >> 16) | (1u << 16)),          // tensor_dim0[31:16], dim1=1
               (int)((elems & 0xFFFFu) << 16),             // tile_dim0 = elems
               1,                                          // tile_dim1=1, tile_dim2=0
               (int)elems,                                 // tensor_dim0_stride[31:0]
               0, 0 };
  i32x4 g2 = { 1, 0, 0, 0 };                               // tensor_dim2=1
  i32x4 g3 = { 0, 0, 0, 0 };
#if __clang_major__ >= 23
  i32x8 g4 = { 0, 0, 0, 0, 0, 0, 0, 0 };
  __builtin_amdgcn_tensor_load_to_lds(g0, g1, g2, g3, g4, 0);
#else
  __builtin_amdgcn_tensor_load_to_lds(g0, g1, g2, g3, 0);
#endif
}

// Stage 16 rows x H_ bf16 (32 KB, contiguous) into LDS via TDM.
__device__ __forceinline__ void stage_rows_tdm(bf16_t* lds, const bf16_t* src,
                                               int wave) {
  if (wave == 0) {
    tdm_load_to_lds((void*)lds, (const void*)src, 16 * H_ * 2);
    __builtin_amdgcn_s_wait_tensorcnt(0);
  }
  __syncthreads();
}

// ---------------------------------------------------------------------------
// Single 16x16 tile of A(16xK) @ W^T with v_wmma_f32_16x16x32_bf16.
// Per-lane layouts per ISA 7.12.2 (A: row=lane%16, K halves by lane/16;
// B: col = W row = n0+lane%16, K 8*half+0..7 then 16+8*half+0..7; D: 8 f32 rows).
// ---------------------------------------------------------------------------
__device__ __forceinline__ f32x8 gemm_tile16(const bf16_t* __restrict__ As,
                                             const bf16_t* __restrict__ W,
                                             int n0, int K, int lane) {
  const int half = lane >> 4;
  const int nl   = lane & 15;
  const bf16_t* ap = As + (size_t)nl * K + half * 8;
  const bf16_t* bp = W + (size_t)(n0 + nl) * K + half * 8;
  f32x8 acc = {};
  for (int k = 0; k < K; k += 32) {
    union { bf16x16 v; uint4 q[2]; } ua, ub;
    ua.q[0] = *(const uint4*)(ap + k);
    ua.q[1] = *(const uint4*)(ap + k + 16);
    ub.q[0] = *(const uint4*)(bp + k);
    ub.q[1] = *(const uint4*)(bp + k + 16);
    __builtin_prefetch((const void*)(bp + k + 512), 0, 3);
    acc = __builtin_amdgcn_wmma_f32_16x16x32_bf16(false, ua.v, false, ub.v,
                                                  (short)0, acc, false, false);
  }
  return acc;
}

// ---------------------------------------------------------------------------
// 4 adjacent 16x16 tiles (16x64 of output) per wave: one A fetch feeds four
// independent WMMA accumulator chains (better XDL pipelining, 4x less LDS
// traffic per FLOP).
// ---------------------------------------------------------------------------
__device__ __forceinline__ void gemm_tile16x4(const bf16_t* __restrict__ As,
                                              const bf16_t* __restrict__ W,
                                              int n0, int K, int lane,
                                              f32x8 acc[4]) {
  const int half = lane >> 4;
  const int nl   = lane & 15;
  const bf16_t* ap  = As + (size_t)nl * K + half * 8;
  const bf16_t* bp0 = W + (size_t)(n0 + nl) * K + half * 8;
  for (int k = 0; k < K; k += 32) {
    union { bf16x16 v; uint4 q[2]; } ua;
    ua.q[0] = *(const uint4*)(ap + k);
    ua.q[1] = *(const uint4*)(ap + k + 16);
#pragma unroll
    for (int j = 0; j < 4; ++j) {
      const bf16_t* bp = bp0 + (size_t)(j * 16) * K;
      union { bf16x16 v; uint4 q[2]; } ub;
      ub.q[0] = *(const uint4*)(bp + k);
      ub.q[1] = *(const uint4*)(bp + k + 16);
      __builtin_prefetch((const void*)(bp + k + 512), 0, 3);
      acc[j] = __builtin_amdgcn_wmma_f32_16x16x32_bf16(false, ua.v, false, ub.v,
                                                       (short)0, acc[j], false, false);
    }
  }
}

// Sense-reversing device-wide barrier (persistent-kernel sync, 32 arrivals).
__device__ __forceinline__ void grid_barrier(int* bar, int nblk) {
  __syncthreads();
  if (threadIdx.x == 0) {
    volatile int* vb = (volatile int*)bar;
    __threadfence();
    int gen = vb[1];
    if (atomicAdd(&bar[0], 1) == nblk - 1) {
      vb[0] = 0;
      __threadfence();
      atomicAdd(&bar[1], 1);
    } else {
      while (vb[1] == gen) __builtin_amdgcn_s_sleep(2);
    }
    __threadfence();
  }
  __syncthreads();
}

// ---------------------------------------------------------------------------
__global__ void __launch_bounds__(TPB) cast_f32_bf16(const float* __restrict__ in,
                                                     bf16_t* __restrict__ out, int n) {
  int i = blockIdx.x * TPB + threadIdx.x;
  if (i < n) out[i] = (bf16_t)in[i];
}

__global__ void __launch_bounds__(TPB) init_state(bf16_t* h_bf, float* c_st,
                                                  int n, int* bar) {
  int i = blockIdx.x * TPB + threadIdx.x;
  if (i < n) { h_bf[i] = (bf16_t)0.0f; c_st[i] = 0.0f; }
  if (i == 0) { bar[0] = 0; bar[1] = 0; }
}

// ---------------------------------------------------------------------------
// Input projection: xin[t][b][g] = x[b][t][:] @ Wi.T + bi  (bf16 out).
// grid = (8, T_*4): y -> (t, 16-row b-tile); 8 waves x 4 tiles -> 512 cols.
// Manual LDS staging here because it fuses the f32->bf16 cast of x.
// ---------------------------------------------------------------------------
__global__ void __launch_bounds__(TPB, 1) proj_kernel(const float* __restrict__ x,
                                                      const bf16_t* __restrict__ Wi,
                                                      const float* __restrict__ bi,
                                                      bf16_t* __restrict__ xin) {
  __shared__ __align__(16) bf16_t ldsA[16 * D_];   // 32 KB
  const int tid = threadIdx.x, lane = tid & 31, wave = tid >> 5;
  const int nl = lane & 15;
  const int mt = blockIdx.y;
  const int t  = mt >> 2;
  const int b0 = (mt & 3) * 16;

  for (int e = tid; e < 16 * D_; e += TPB) {
    int r = e >> 10, c = e & (D_ - 1);
    ldsA[e] = (bf16_t)x[((size_t)(b0 + r) * T_ + t) * D_ + c];
  }
  __syncthreads();

  const int nbase = (blockIdx.x * 8 + wave) * 64;
  f32x8 acc[4] = {};
  gemm_tile16x4(ldsA, Wi, nbase, D_, lane, acc);

  const int mb = b0 + ((lane >> 4) << 3);
  bf16_t* xt = xin + (size_t)t * (B_ * G4_);
#pragma unroll
  for (int j = 0; j < 4; ++j) {
    const int n = nbase + j * 16 + nl;
    const float bin = bi[n];
#pragma unroll
    for (int r = 0; r < 8; ++r)
      xt[(mb + r) * G4_ + n] = (bf16_t)(acc[j][r] + bin);
  }
}

// ---------------------------------------------------------------------------
// Persistent recurrent kernel: 32 blocks x 256 threads; all blocks active in
// all 3 phases; A-tiles staged via the Tensor Data Mover.
//   phase 1: preact = xin[t] + h @ Wh.T + bh  -> sigmoid gates / g-preact
//   phase 2: g = pA @ Wa.T + ba; c = c*f + i*g
//   phase 3: h = o * (c @ Wa.T + ba)
// ---------------------------------------------------------------------------
__global__ void __launch_bounds__(TPB, 1) lstm_recurrent(
    const bf16_t* __restrict__ xin, const bf16_t* __restrict__ Wh,
    const float* __restrict__ bh,   const bf16_t* __restrict__ Wa,
    const float* __restrict__ ba,
    float* __restrict__ ig, float* __restrict__ fg, float* __restrict__ og,
    bf16_t* __restrict__ pA, bf16_t* __restrict__ cA,
    float* __restrict__ c_st, bf16_t* __restrict__ h_bf,
    float* __restrict__ out, int* bar) {
  __shared__ __align__(16) bf16_t ldsA[16 * H_];   // 32 KB, reused per phase

  const int tid = threadIdx.x, lane = tid & 31, wave = tid >> 5;
  const int nl  = lane & 15;
  const int blk = blockIdx.x;

  // All phases share the same 16-row m-tile per block.
  const int m0 = (blk >> 3) * 16;                    // 0..3
  const int mb = m0 + ((lane >> 4) << 3);
  // Phase 1: each wave 4 n-tiles (16x64): 8 blk * 8 waves * 64 = 4096 cols per m-tile.
  const int nbase1 = (blk & 7) * 512 + wave * 64;
  // Phases 2/3: each wave one n-tile: 32 blk * 8 waves * 16 = ... (4 m x 64 n).
  const int n0_23 = ((blk & 7) * 8 + wave) * 16;

  for (int t = 0; t < T_; ++t) {
    // ---- phase 1 ----
    stage_rows_tdm(ldsA, h_bf + m0 * H_, wave);      // tensor_load_to_lds
    {
      f32x8 acc[4] = {};
      gemm_tile16x4(ldsA, Wh, nbase1, H_, lane, acc);
      const bf16_t* xt = xin + (size_t)t * (B_ * G4_);
      const int region = nbase1 >> 10;               // uniform per wave
      if (region < 3) {
        float* gp = (region == 0) ? ig : ((region == 1) ? fg : og);
#pragma unroll
        for (int j = 0; j < 4; ++j) {
          const int n = nbase1 + j * 16 + nl;
          const int nn = n & (H_ - 1);
          const float bhn = bh[n];
#pragma unroll
          for (int r = 0; r < 8; ++r) {
            const int m = mb + r;
            float v = acc[j][r] + bhn + (float)xt[m * G4_ + n];
            gp[m * H_ + nn] = 1.0f / (1.0f + __expf(-v));
          }
        }
      } else {
#pragma unroll
        for (int j = 0; j < 4; ++j) {
          const int n = nbase1 + j * 16 + nl;
          const int nn = n - 3 * H_;
          const float bhn = bh[n];
#pragma unroll
          for (int r = 0; r < 8; ++r) {
            const int m = mb + r;
            float v = acc[j][r] + bhn + (float)xt[m * G4_ + n];
            pA[m * H_ + nn] = (bf16_t)v;
          }
        }
      }
    }
    grid_barrier(bar, NBLK);

    // ---- phase 2 ----
    stage_rows_tdm(ldsA, pA + m0 * H_, wave);
    {
      f32x8 acc = gemm_tile16(ldsA, Wa, n0_23, H_, lane);
      const int n = n0_23 + nl;
      const float ban = ba[n];
#pragma unroll
      for (int r = 0; r < 8; ++r) {
        const int idx = (mb + r) * H_ + n;
        float g  = acc[r] + ban;
        float cn = c_st[idx] * fg[idx] + ig[idx] * g;
        c_st[idx] = cn;
        cA[idx]   = (bf16_t)cn;
      }
    }
    grid_barrier(bar, NBLK);

    // ---- phase 3 ----
    stage_rows_tdm(ldsA, cA + m0 * H_, wave);
    {
      f32x8 acc = gemm_tile16(ldsA, Wa, n0_23, H_, lane);
      const int n = n0_23 + nl;
      const float ban = ba[n];
#pragma unroll
      for (int r = 0; r < 8; ++r) {
        const int idx = (mb + r) * H_ + n;
        float ht = og[idx] * (acc[r] + ban);
        h_bf[idx] = (bf16_t)ht;
        if (t == T_ - 1) out[idx] = ht;              // final h_T -> d_out (1,B,H)
      }
    }
    grid_barrier(bar, NBLK);
  }
}

// ---------------------------------------------------------------------------
extern "C" void kernel_launch(void* const* d_in, const int* in_sizes, int n_in,
                              void* d_out, int out_size, void* d_ws, size_t ws_size,
                              hipStream_t stream) {
  const float* x  = (const float*)d_in[0];   // (B,T,D)
  const float* Wi = (const float*)d_in[1];   // (4H,D)
  const float* bi = (const float*)d_in[2];   // (4H)
  const float* Wh = (const float*)d_in[3];   // (4H,H)
  const float* bh = (const float*)d_in[4];   // (4H)
  const float* Wa = (const float*)d_in[5];   // (H,H)
  const float* ba = (const float*)d_in[6];   // (H)
  float* out = (float*)d_out;                // (1,B,H) f32

  char* ws = (char*)d_ws;
  size_t off = 0;
  auto alloc = [&](size_t bytes) -> void* {
    void* p = ws + off;
    off += (bytes + 255) & ~(size_t)255;
    return p;
  };
  bf16_t* Wi_bf = (bf16_t*)alloc((size_t)G4_ * D_ * 2);      //   8 MB
  bf16_t* Wh_bf = (bf16_t*)alloc((size_t)G4_ * H_ * 2);      //   8 MB
  bf16_t* Wa_bf = (bf16_t*)alloc((size_t)H_ * H_ * 2);       //   2 MB
  bf16_t* xin   = (bf16_t*)alloc((size_t)T_ * B_ * G4_ * 2); // 256 MB
  bf16_t* pA    = (bf16_t*)alloc((size_t)B_ * H_ * 2);
  bf16_t* cA    = (bf16_t*)alloc((size_t)B_ * H_ * 2);
  bf16_t* h_bf  = (bf16_t*)alloc((size_t)B_ * H_ * 2);
  float*  c_st  = (float*)alloc((size_t)B_ * H_ * 4);
  float*  ig    = (float*)alloc((size_t)B_ * H_ * 4);
  float*  fg    = (float*)alloc((size_t)B_ * H_ * 4);
  float*  og    = (float*)alloc((size_t)B_ * H_ * 4);
  int*    bar   = (int*)alloc(256);

  cast_f32_bf16<<<(G4_ * D_ + TPB - 1) / TPB, TPB, 0, stream>>>(Wi, Wi_bf, G4_ * D_);
  cast_f32_bf16<<<(G4_ * H_ + TPB - 1) / TPB, TPB, 0, stream>>>(Wh, Wh_bf, G4_ * H_);
  cast_f32_bf16<<<(H_ * H_ + TPB - 1) / TPB, TPB, 0, stream>>>(Wa, Wa_bf, H_ * H_);
  init_state<<<(B_ * H_ + TPB - 1) / TPB, TPB, 0, stream>>>(h_bf, c_st, B_ * H_, bar);

  // Input projection GEMM (32768 x 4096 x 1024), WMMA bf16, 16x64 per wave.
  proj_kernel<<<dim3(8, T_ * 4), TPB, 0, stream>>>(x, Wi_bf, bi, xin);

  // Persistent recurrent scan: 32 blocks, TDM staging, device-wide barriers.
  lstm_recurrent<<<NBLK, TPB, 0, stream>>>(xin, Wh_bf, bh, Wa_bf, ba,
                                           ig, fg, og, pA, cA, c_st, h_bf,
                                           out, bar);
}
